// Plant_Identifier_Custom_20942260535374
// MI455X (gfx1250) — matrix operand order repaired and checked by reading, not compiled, not run on hardware
//
#include <hip/hip_runtime.h>
#include <hip/hip_bf16.h>

typedef __attribute__((ext_vector_type(16))) _Float16 v16h;
typedef __attribute__((ext_vector_type(8)))  float    v8f;

// ---------------------------------------------------------------------------
// Utility: zero a float region (stats + fc1 split-K accumulator)
// ---------------------------------------------------------------------------
__global__ void zero_f32(float* __restrict__ p, int n) {
    int i = blockIdx.x * 256 + threadIdx.x;
    if (i < n) p[i] = 0.f;
}

// ---------------------------------------------------------------------------
// conv1: [64,3,224,224] fp32, 2x2 kernel, pad=1 -> [64,16,225,225] f16 raw
// Also accumulates per-channel sum / sumsq for BN (LDS reduce + global atomic)
// ---------------------------------------------------------------------------
__global__ void __launch_bounds__(256)
conv1_2x2(const float* __restrict__ d, const float* __restrict__ w,
          const float* __restrict__ bias, _Float16* __restrict__ out,
          float* __restrict__ stats) {
    __shared__ float s_sum[16], s_sq[16];
    if (threadIdx.x < 16) { s_sum[threadIdx.x] = 0.f; s_sq[threadIdx.x] = 0.f; }
    __syncthreads();

    const long long total = 64LL * 16 * 225 * 225;
    long long i = (long long)blockIdx.x * 256 + threadIdx.x;
    if (i < total) {
        int x = (int)(i % 225); long long t = i / 225;
        int y = (int)(t % 225); t /= 225;
        int o = (int)(t % 16);
        int b = (int)(t / 16);
        float acc = bias[o];
        #pragma unroll
        for (int c = 0; c < 3; ++c)
            #pragma unroll
            for (int r = 0; r < 2; ++r) {
                int yy = y + r - 1;
                if (yy < 0 || yy >= 224) continue;
                #pragma unroll
                for (int s = 0; s < 2; ++s) {
                    int xx = x + s - 1;
                    if (xx < 0 || xx >= 224) continue;
                    acc += d[(((size_t)b * 3 + c) * 224 + yy) * 224 + xx] *
                           w[((o * 3 + c) * 2 + r) * 2 + s];
                }
            }
        out[i] = (_Float16)acc;
        atomicAdd(&s_sum[o], acc);
        atomicAdd(&s_sq[o], acc * acc);
    }
    __syncthreads();
    if (threadIdx.x < 16) {
        atomicAdd(&stats[threadIdx.x],      s_sum[threadIdx.x]);
        atomicAdd(&stats[16 + threadIdx.x], s_sq[threadIdx.x]);
    }
}

// ---------------------------------------------------------------------------
// Fused BatchNorm(train) + ReLU + 2x2/2 maxpool (VALID). f16 in -> f16 out.
// ---------------------------------------------------------------------------
__global__ void __launch_bounds__(256)
bn_relu_pool(const _Float16* __restrict__ in, _Float16* __restrict__ out,
             const float* __restrict__ stats, const float* __restrict__ gamma,
             const float* __restrict__ beta, int C, int Hin, int Win,
             int Hout, int Wout, float invCnt, long long total) {
    long long i = (long long)blockIdx.x * 256 + threadIdx.x;
    if (i >= total) return;
    int wo = (int)(i % Wout); long long t = i / Wout;
    int ho = (int)(t % Hout); t /= Hout;
    int c  = (int)(t % C);
    int b  = (int)(t / C);
    float mean  = stats[c] * invCnt;
    float var   = stats[C + c] * invCnt - mean * mean;
    float scale = gamma[c] * rsqrtf(var + 1e-5f);
    float shift = beta[c] - mean * scale;
    const _Float16* base = in + (((size_t)b * C + c) * Hin + 2 * ho) * Win + 2 * wo;
    float v00 = fmaxf((float)base[0]       * scale + shift, 0.f);
    float v01 = fmaxf((float)base[1]       * scale + shift, 0.f);
    float v10 = fmaxf((float)base[Win]     * scale + shift, 0.f);
    float v11 = fmaxf((float)base[Win + 1] * scale + shift, 0.f);
    out[i] = (_Float16)fmaxf(fmaxf(v00, v01), fmaxf(v10, v11));
}

// ---------------------------------------------------------------------------
// Convert fp32 weight [N][K] (row-major) into WMMA-fragment-packed f16 B:
// for each 32-row K-chunk, the 32 halves of column n are stored contiguously:
//   out[(chunk*N + n)*32 + t] = in[n][chunk*32 + t]   (zero-padded K -> Kpad)
// A lane then reads its 16 B-fragment halves as one contiguous 32-byte run
// (kbB = 0 or 16), which the compiler vectorizes into 2x global_load_b128.
// Works for OIHW conv weights (k = c*9+r*3+s matches memory order) and fc1_w.
// ---------------------------------------------------------------------------
__global__ void __launch_bounds__(256)
pack_b_f16(const float* __restrict__ in, _Float16* __restrict__ out,
           int N, int K, long long total) {
    long long i = (long long)blockIdx.x * 256 + threadIdx.x;
    if (i >= total) return;
    int t = (int)(i & 31);
    long long q = i >> 5;
    int n = (int)(q % N);
    long long chunk = q / N;
    long long k = chunk * 32 + t;
    out[i] = (k < K) ? (_Float16)in[(size_t)n * K + k] : (_Float16)0.f;
}

// ---------------------------------------------------------------------------
// Implicit-GEMM 3x3/pad1 conv via V_WMMA_F32_16X16X32_F16.
// One wave computes a 16-pixel (M) x Cout (N = NT*16) tile.
// A fragment (16x32 f16): lane<16 holds K {0..15}, lane>=16 holds {16..31}.
// B fragment: contiguous 16 halves from the chunk-packed weight buffer.
// Epilogue adds bias, stores raw f16 conv output, reduces BN sum/sumsq.
// ---------------------------------------------------------------------------
template<int NT>
__global__ void __launch_bounds__(256)
conv3x3_wmma(const _Float16* __restrict__ in, const _Float16* __restrict__ Bw,
             const float* __restrict__ bias, _Float16* __restrict__ out,
             float* __restrict__ stats,
             int Cin, int H, int W, int Kpad, int tilesPerImg) {
    constexpr int Cout = NT * 16;
    __shared__ float s_sum[Cout], s_sq[Cout];
    const int tid = threadIdx.x;
    if (tid < Cout) { s_sum[tid] = 0.f; s_sq[tid] = 0.f; }
    __syncthreads();

    const int wave   = tid >> 5;          // wave32
    const int lane   = tid & 31;
    const int tileId = blockIdx.x * 8 + wave;
    const int img    = tileId / tilesPerImg;
    const int m0     = (tileId % tilesPerImg) * 16;

    const int ml  = lane & 15;
    const int p   = m0 + ml;
    const int hp  = p / W;
    const int wp  = p % W;
    const int kbA = (lane < 16) ? 0 : 16;
    const int kbB = (lane < 16) ? 0 : 16;
    const int nlo = lane & 15;
    const int Ktot = Cin * 9;
    const _Float16* inImg = in + (size_t)img * Cin * H * W;

    v8f acc[NT];
    #pragma unroll
    for (int t = 0; t < NT; ++t) acc[t] = (v8f){};

    for (int kc = 0; kc < Kpad; kc += 32) {
        v16h a{};
        #pragma unroll
        for (int j = 0; j < 16; ++j) {
            int k = kc + kbA + j;
            _Float16 v = (_Float16)0.f;
            if (k < Ktot) {
                int c  = k / 9;
                int rs = k - c * 9;
                int r  = rs / 3;
                int s  = rs - r * 3;
                int y  = hp + r - 1;
                int x  = wp + s - 1;
                if (y >= 0 && y < H && x >= 0 && x < W)
                    v = inImg[((size_t)c * H + y) * W + x];
            }
            a[j] = v;
        }
        #pragma unroll
        for (int t = 0; t < NT; ++t) {
            const int n = t * 16 + nlo;
            const _Float16* bbase =
                Bw + ((size_t)(kc >> 5) * Cout + n) * 32 + kbB;
            v16h b;
            #pragma unroll
            for (int j = 0; j < 16; ++j)
                b[j] = bbase[j];
            acc[t] = __builtin_amdgcn_wmma_f32_16x16x32_f16(
                false, a, false, b, (short)0, acc[t], false, false);
        }
    }

    // Epilogue: C/D layout -> lane row m = r + (lane<16 ? 0 : 8), col n = lane%16
    const int mOff = (lane < 16) ? 0 : 8;
    #pragma unroll
    for (int t = 0; t < NT; ++t) {
        const int n = t * 16 + nlo;
        const float bs = bias[n];
        float lsum = 0.f, lsq = 0.f;
        #pragma unroll
        for (int r = 0; r < 8; ++r) {
            int pp = m0 + mOff + r;
            int hh = pp / W, ww = pp % W;
            float v = acc[t][r] + bs;
            out[(((size_t)img * Cout + n) * H + hh) * W + ww] = (_Float16)v;
            lsum += v; lsq += v * v;
        }
        atomicAdd(&s_sum[n], lsum);
        atomicAdd(&s_sq[n], lsq);
    }
    __syncthreads();
    if (tid < Cout) {
        atomicAdd(&stats[tid],        s_sum[tid]);
        atomicAdd(&stats[Cout + tid], s_sq[tid]);
    }
}

// ---------------------------------------------------------------------------
// fc1: [64,50176] f16 x [50176,256] f16 -> fp32, split-K WMMA.
// K = 50176 = 32 splits x 49 chunks x 32. Results atomically reduced.
// A: 2x b128 per lane (contiguous); B: 2x b128 per lane (chunk-packed).
// ---------------------------------------------------------------------------
__global__ void __launch_bounds__(256)
fc1_wmma(const _Float16* __restrict__ A, const _Float16* __restrict__ Bw,
         float* __restrict__ accOut) {
    const int tid  = threadIdx.x;
    const int wave = tid >> 5;
    const int lane = tid & 31;
    const int wg   = blockIdx.x * 8 + wave;   // 2048 wave-tiles total
    const int mt   = wg & 3;                   // 4 M-tiles (64 rows)
    const int nt   = (wg >> 2) & 15;           // 16 N-tiles (256 cols)
    const int sp   = wg >> 6;                  // 32 K-splits

    const int kbA = (lane < 16) ? 0 : 16;
    const int kbB = (lane < 16) ? 0 : 16;
    const int n   = nt * 16 + (lane & 15);
    const _Float16* Arow  = A + (size_t)(mt * 16 + (lane & 15)) * 50176;
    const int chunk0 = sp * 49;                // first K-chunk of this split

    v8f acc = {};
    for (int ch = 0; ch < 49; ++ch) {
        const int kc = (chunk0 + ch) * 32;
        const _Float16* abase = Arow + kc + kbA;
        const _Float16* bbase = Bw + ((size_t)(chunk0 + ch) * 256 + n) * 32 + kbB;
        // keep HBM ahead of the single-pass streaming consumption
        __builtin_prefetch(abase + 32, 0, 3);
        __builtin_prefetch(bbase + 256 * 32, 0, 3);
        v16h a, b;
        #pragma unroll
        for (int j = 0; j < 16; ++j) {
            a[j] = abase[j];
            b[j] = bbase[j];
        }
        acc = __builtin_amdgcn_wmma_f32_16x16x32_f16(
            false, a, false, b, (short)0, acc, false, false);
    }
    const int mBase = mt * 16 + ((lane < 16) ? 0 : 8);
    #pragma unroll
    for (int r = 0; r < 8; ++r)
        atomicAdd(&accOut[(size_t)(mBase + r) * 256 + n], acc[r]);
}

// ---------------------------------------------------------------------------
// Tail: relu(fc1acc + fc1_b) @ fc2_w.T + fc2_b  ->  d_out [64,5] fp32
// ---------------------------------------------------------------------------
__global__ void __launch_bounds__(256)
fc_tail(const float* __restrict__ acc1, const float* __restrict__ fc1b,
        const float* __restrict__ fc2w, const float* __restrict__ fc2b,
        float* __restrict__ outp) {
    int t = blockIdx.x * 256 + threadIdx.x;
    if (t >= 64 * 5) return;
    int m = t / 5, n = t % 5;
    float s = fc2b[n];
    for (int j = 0; j < 256; ++j) {
        float a = fmaxf(acc1[m * 256 + j] + fc1b[j], 0.f);
        s += a * fc2w[n * 256 + j];
    }
    outp[t] = s;
}

// ---------------------------------------------------------------------------
// Launch
// ---------------------------------------------------------------------------
extern "C" void kernel_launch(void* const* d_in, const int* in_sizes, int n_in,
                              void* d_out, int out_size, void* d_ws, size_t ws_size,
                              hipStream_t stream) {
    (void)in_sizes; (void)n_in; (void)out_size; (void)ws_size;

    const float* d       = (const float*)d_in[0];
    const float* conv1_w = (const float*)d_in[1];
    const float* conv1_b = (const float*)d_in[2];
    const float* bn1_g   = (const float*)d_in[3];
    const float* bn1_b   = (const float*)d_in[4];
    const float* conv2_w = (const float*)d_in[5];
    const float* conv2_b = (const float*)d_in[6];
    const float* bn2_g   = (const float*)d_in[7];
    const float* bn2_b   = (const float*)d_in[8];
    const float* conv3_w = (const float*)d_in[9];
    const float* conv3_b = (const float*)d_in[10];
    const float* bn3_g   = (const float*)d_in[11];
    const float* bn3_b   = (const float*)d_in[12];
    const float* fc1_w   = (const float*)d_in[13];
    const float* fc1_b   = (const float*)d_in[14];
    const float* fc2_w   = (const float*)d_in[15];
    const float* fc2_b   = (const float*)d_in[16];
    float* outp = (float*)d_out;

    // ---- carve workspace (256B aligned) ----
    char* p = (char*)d_ws;
    auto alloc = [&](size_t bytes) -> void* {
        void* r = (void*)p;
        p += (bytes + 255) & ~(size_t)255;
        return r;
    };
    // contiguous fp32 accumulators that must be zeroed each launch:
    // stats1[32] | stats2[64] | stats3[128] | fc1acc[16384]  = 16608 floats
    float* statsBase = (float*)alloc(16608 * sizeof(float));
    float* stats1 = statsBase;
    float* stats2 = statsBase + 32;
    float* stats3 = statsBase + 96;
    float* fc1acc = statsBase + 224;

    _Float16* conv1raw = (_Float16*)alloc((size_t)64 * 16 * 225 * 225 * 2); // 103.7 MB
    _Float16* pool1    = (_Float16*)alloc((size_t)64 * 16 * 112 * 112 * 2); //  25.7 MB
    _Float16* conv2raw = (_Float16*)alloc((size_t)64 * 32 * 112 * 112 * 2); //  51.4 MB
    _Float16* pool2    = (_Float16*)alloc((size_t)64 * 32 * 56 * 56 * 2);   //  12.8 MB
    _Float16* conv3raw = (_Float16*)alloc((size_t)64 * 64 * 56 * 56 * 2);   //  25.7 MB
    _Float16* pool3    = (_Float16*)alloc((size_t)64 * 64 * 28 * 28 * 2);   //   6.4 MB
    _Float16* w2f16    = (_Float16*)alloc((size_t)160 * 32 * 2);            // 5 chunks x 32ch
    _Float16* w3f16    = (_Float16*)alloc((size_t)288 * 64 * 2);            // 9 chunks x 64ch
    _Float16* fc1wf16  = (_Float16*)alloc((size_t)50176 * 256 * 2);         //  25.7 MB

    // 0) zero BN stats + fc1 split-K accumulator (idempotent per launch)
    zero_f32<<<(16608 + 255) / 256, 256, 0, stream>>>(statsBase, 16608);

    // 1) conv1 + BN stats.  64*16*225*225 = 51,840,000 = 202,500 * 256
    conv1_2x2<<<202500, 256, 0, stream>>>(d, conv1_w, conv1_b, conv1raw, stats1);

    // 2) BN+ReLU+pool -> [64,16,112,112] f16
    bn_relu_pool<<<50176, 256, 0, stream>>>(conv1raw, pool1, stats1, bn1_g, bn1_b,
                                            16, 225, 225, 112, 112,
                                            1.f / 3240000.f, 12845056LL);

    // 3) conv2 weights -> chunk-packed f16 B (Kpad=160, 5 chunks x N=32 x 32)
    pack_b_f16<<<20, 256, 0, stream>>>(conv2_w, w2f16, 32, 144, 5120LL);

    // 4) conv2 implicit-GEMM WMMA: 64 imgs * 784 tiles = 50,176 waves = 6272 blocks
    conv3x3_wmma<2><<<6272, 256, 0, stream>>>(pool1, w2f16, conv2_b, conv2raw,
                                              stats2, 16, 112, 112, 160, 784);

    // 5) BN+ReLU+pool -> [64,32,56,56] f16
    bn_relu_pool<<<25088, 256, 0, stream>>>(conv2raw, pool2, stats2, bn2_g, bn2_b,
                                            32, 112, 112, 56, 56,
                                            1.f / 802816.f, 6422528LL);

    // 6) conv3 weights -> chunk-packed f16 B (Kpad=288, 9 chunks x N=64 x 32)
    pack_b_f16<<<72, 256, 0, stream>>>(conv3_w, w3f16, 64, 288, 18432LL);

    // 7) conv3 implicit-GEMM WMMA: 64 imgs * 196 tiles = 12,544 waves = 1568 blocks
    conv3x3_wmma<4><<<1568, 256, 0, stream>>>(pool2, w3f16, conv3_b, conv3raw,
                                              stats3, 32, 56, 56, 288, 196);

    // 8) BN+ReLU+pool -> [64,64,28,28] f16 (flattened = fc1 activation)
    bn_relu_pool<<<12544, 256, 0, stream>>>(conv3raw, pool3, stats3, bn3_g, bn3_b,
                                            64, 56, 56, 28, 28,
                                            1.f / 200704.f, 3211264LL);

    // 9) fc1 weights -> chunk-packed f16 B (1568 chunks x N=256 x 32)
    pack_b_f16<<<50176, 256, 0, stream>>>(fc1_w, fc1wf16, 256, 50176, 12845056LL);

    // 10) fc1 split-K WMMA: 4 Mt * 16 Nt * 32 splits = 2048 waves = 256 blocks
    fc1_wmma<<<256, 256, 0, stream>>>(pool3, fc1wf16, fc1acc);

    // 11) bias + relu + fc2 -> [64,5]
    fc_tail<<<2, 256, 0, stream>>>(fc1acc, fc1_b, fc2_w, fc2_b, outp);
}
